// SGNN_6090263625849
// MI455X (gfx1250) — matrix operand order repaired and checked by reading, not compiled, not run on hardware
//
#include <hip/hip_runtime.h>
#include <hip/hip_bf16.h>

#define VD 32
#define HD 64
#define ES 68
#define NN 10000
#define EE 80000
#define NET_KPAD 1248   // 39 K-tiles (1220 real, zero padded)
#define SELF_KPAD 1152  // 36 K-tiles exactly
#define OUTD 1088       // 68 N-tiles
#define EMB1_KPAD 96    // 3 K-tiles (65 real)
#define EMB2_KPAD 64    // 2 K-tiles exactly

typedef __attribute__((ext_vector_type(16))) __bf16 v16bf;
typedef __attribute__((ext_vector_type(8)))  float  v8f;

// float -> bf16 (RNE) via hardware convert (v_cvt_*bf16_f32 on gfx1250)
__device__ __forceinline__ unsigned short f2bf(float x) {
  __bf16 h = (__bf16)x;
  return __builtin_bit_cast(unsigned short, h);
}

__device__ __forceinline__ void gadd(float* p, float v) {
  __hip_atomic_fetch_add(p, v, __ATOMIC_RELAXED, __HIP_MEMORY_SCOPE_AGENT);
}

// A fragment: 16(M) x 32(K) bf16 from row-major LDS [16][stride] ushort.
__device__ __forceinline__ v16bf frag_a(const unsigned short* A, int stride, int lane, int kt) {
  union { unsigned short u[16]; v16bf v; } a;
  const unsigned short* row = A + (lane & 15) * stride + kt * 32;
  int hi = ((lane >> 4) & 1) * 8;
#pragma unroll
  for (int v = 0; v < 8; ++v) {
    int k = ((v >= 4) ? 16 : 0) + hi + ((v & 3) << 1);
    a.u[2 * v]     = row[k];
    a.u[2 * v + 1] = row[k + 1];
  }
  return a.v;
}

// B fragment: pre-packed 512 ushorts per (kt,nt) tile; lane reads 16 contiguous ushorts (32B)
__device__ __forceinline__ v16bf frag_b(const unsigned short* tile, int lane) {
  union { unsigned short u[16]; v16bf v; } b;
  const unsigned short* p = tile + lane * 16;
#pragma unroll
  for (int e = 0; e < 16; ++e) b.u[e] = p[e];
  return b.v;
}

__device__ __forceinline__ v8f mm_tiles(const unsigned short* A, int stride,
                                        const unsigned short* Bp, int ntTot, int nt,
                                        int kTiles, int lane) {
  v8f acc = {0.f, 0.f, 0.f, 0.f, 0.f, 0.f, 0.f, 0.f};
  for (int kt = 0; kt < kTiles; ++kt) {
    if (kt + 1 < kTiles)
      __builtin_prefetch(Bp + (size_t)((kt + 1) * ntTot + nt) * 512, 0, 0);
    v16bf a = frag_a(A, stride, lane, kt);
    v16bf b = frag_b(Bp + (size_t)(kt * ntTot + nt) * 512, lane);
    acc = __builtin_amdgcn_wmma_f32_16x16x32_bf16(false, a, false, b, (short)0, acc,
                                                  false, false);
  }
  return acc;
}

__global__ void k_zero(float* p, int n) {
  int i = blockIdx.x * blockDim.x + threadIdx.x;
  for (; i < n; i += gridDim.x * blockDim.x) p[i] = 0.f;
}

// Pack f32 [K][Ncols] weights into WMMA-B tiles of bf16, K zero-padded to KpadTiles*32.
__global__ void k_pack(const float* __restrict__ src, unsigned short* __restrict__ dst,
                       int K, int KpadTiles, int Ncols) {
  int total = KpadTiles * (Ncols >> 4) * 512;
  int gi = blockIdx.x * blockDim.x + threadIdx.x;
  if (gi >= total) return;
  int e = gi & 15, lane = (gi >> 4) & 31, tile = gi >> 9;
  int ntTot = Ncols >> 4;
  int kt = tile / ntTot, nt = tile % ntTot;
  int k = kt * 32 + (e & 7) + ((e >> 3) << 4) + ((lane >> 4) & 1) * 8;
  int n = nt * 16 + (lane & 15);
  dst[gi] = (k < K) ? f2bf(src[(size_t)k * Ncols + n]) : (unsigned short)0;
}

// ---------------- edge kernel: 16 edges / block, 256 threads (8 waves) ----------------
__global__ __launch_bounds__(256) void k_edge(
    const float* __restrict__ f, const float* __restrict__ s,
    const float* __restrict__ edge_f, const float* __restrict__ edge_s,
    const unsigned short* __restrict__ E1p,
    const float* __restrict__ b1, const float* __restrict__ b2, const float* __restrict__ b3,
    const unsigned short* __restrict__ W1p, const unsigned short* __restrict__ W2p,
    const unsigned short* __restrict__ W3p,
    const int* __restrict__ eidx,
    float* __restrict__ fc_sum, float* __restrict__ sc_sum, float* __restrict__ cnt) {
  __shared__ unsigned short sX[16 * NET_KPAD];   // low region reused for 48x96 embed A-panel
  __shared__ float sF[16][3][VD];
  __shared__ unsigned short sH1[16 * HD];
  __shared__ unsigned short sH2[16 * HD];
  __shared__ float sFC[16][3][VD];
  __shared__ float sInvFn[16];
  __shared__ float sSumSq[16];
  __shared__ int sE0[16], sE1[16];

  const int tid = threadIdx.x, lane = tid & 31, wave = tid >> 5;
  const int ebase = blockIdx.x * 16;

  if (tid < 16) {
    sE0[tid] = eidx[ebase + tid];
    sE1[tid] = eidx[EE + ebase + tid];
    sSumSq[tid] = 0.f;
  }
  for (int i = tid; i < 16 * 3 * VD; i += 256) ((float*)sFC)[i] = 0.f;
  __syncthreads();

  // stage embed A-panel: 48 rows (r = e*3+i) x 96 cols = [f[e0](32), f[e1](32), edge_f(1), 0...]
  for (int idx = tid; idx < 48 * EMB1_KPAD; idx += 256) {
    int r = idx / EMB1_KPAD, k = idx % EMB1_KPAD, e = r / 3, i = r % 3;
    float v = 0.f;
    if (k < 32)       v = f[(size_t)sE0[e] * 96 + i * VD + k];
    else if (k < 64)  v = f[(size_t)sE1[e] * 96 + i * VD + (k - 32)];
    else if (k == 64) v = edge_f[(size_t)(ebase + e) * 3 + i];
    sX[idx] = f2bf(v);
  }
  __syncthreads();

  // embed GEMM: [48,96] @ [96,32] -> sF (f32). 6 (mtile,nt) pairs on waves 0..5
  if (wave < 6) {
    int mtile = wave >> 1, nt = wave & 1;
    v8f acc = mm_tiles(sX + mtile * 16 * EMB1_KPAD, EMB1_KPAD, E1p, 2, nt, 3, lane);
    int c = nt * 16 + (lane & 15);
#pragma unroll
    for (int r = 0; r < 8; ++r) {
      int grow = mtile * 16 + r + 8 * ((lane >> 4) & 1);
      sF[grow / 3][grow % 3][c] = acc[r];
    }
  }
  __syncthreads();

  // f2s -> bf16 X[:,0:1024] (+norm); scalar tail cols 1024..1247
  {
    int e = tid >> 4, base = (tid & 15) * 64;
    float ss = 0.f;
    for (int t = 0; t < 64; ++t) {
      int jk = base + t, j = jk >> 5, k = jk & 31;
      float v = sF[e][0][j] * sF[e][0][k] + sF[e][1][j] * sF[e][1][k] +
                sF[e][2][j] * sF[e][2][k];
      ss += v * v;
      sX[e * NET_KPAD + jk] = f2bf(v);
    }
    atomicAdd(&sSumSq[e], ss);
  }
  for (int idx = tid; idx < 16 * 224; idx += 256) {
    int e = idx / 224, o = idx % 224;
    float v = 0.f;
    if (o < 64)        v = s[(size_t)sE0[e] * HD + o];
    else if (o < 128)  v = s[(size_t)sE1[e] * HD + (o - 64)];
    else if (o < 196)  v = edge_s[(size_t)(ebase + e) * ES + (o - 128)];
    sX[e * NET_KPAD + 1024 + o] = f2bf(v);
  }
  __syncthreads();
  if (tid < 16) sInvFn[tid] = 1.f / (sqrtf(sSumSq[tid]) + 1.f);
  __syncthreads();

  // layer 1: [16,1248] @ [1248,64]
  if (wave < 4) {
    v8f acc = mm_tiles(sX, NET_KPAD, W1p, 4, wave, NET_KPAD / 32, lane);
    int n = wave * 16 + (lane & 15);
    float bb = b1[n];
#pragma unroll
    for (int r = 0; r < 8; ++r) {
      int m = r + 8 * ((lane >> 4) & 1);
      float h = acc[r] + bb;
      sH1[m * HD + n] = f2bf(h > 0.f ? h : 0.f);
    }
  }
  __syncthreads();
  // layer 2: [16,64] @ [64,64]
  if (wave < 4) {
    v8f acc = mm_tiles(sH1, HD, W2p, 4, wave, 2, lane);
    int n = wave * 16 + (lane & 15);
    float bb = b2[n];
#pragma unroll
    for (int r = 0; r < 8; ++r) {
      int m = r + 8 * ((lane >> 4) & 1);
      float h = acc[r] + bb;
      sH2[m * HD + n] = f2bf(h > 0.f ? h : 0.f);
    }
  }
  __syncthreads();

  // layer 3: [16,64] @ [64,1088]
  // Matrix cols (tiles nt=2j+h): lane's k = 16h + (lane&15) is fixed -> accumulate f_c in
  // registers over the j-slice, commit with 24 ds atomics per lane. Waves 6,7: scalar tiles.
  if (wave < 6) {
    int h = wave & 1, slice = wave >> 1;
    int j0 = (slice == 0) ? 0 : (slice == 1) ? 11 : 22;
    int j1 = (slice == 0) ? 11 : (slice == 1) ? 22 : 32;
    float fcacc[8][3];
#pragma unroll
    for (int r = 0; r < 8; ++r)
      for (int i = 0; i < 3; ++i) fcacc[r][i] = 0.f;
    for (int j = j0; j < j1; ++j) {
      int nt = 2 * j + h;
      v8f acc = mm_tiles(sH2, HD, W3p, 68, nt, 2, lane);
      float bb = b3[nt * 16 + (lane & 15)];
#pragma unroll
      for (int r = 0; r < 8; ++r) {
        int m = r + 8 * ((lane >> 4) & 1);
        float val = (acc[r] + bb) * sInvFn[m];
        fcacc[r][0] += sF[m][0][j] * val;
        fcacc[r][1] += sF[m][1][j] * val;
        fcacc[r][2] += sF[m][2][j] * val;
      }
    }
    int k = h * 16 + (lane & 15);
#pragma unroll
    for (int r = 0; r < 8; ++r) {
      int m = r + 8 * ((lane >> 4) & 1);
      atomicAdd(&sFC[m][0][k], fcacc[r][0]);
      atomicAdd(&sFC[m][1][k], fcacc[r][1]);
      atomicAdd(&sFC[m][2][k], fcacc[r][2]);
    }
  } else {
    for (int t = 0; t < 2; ++t) {
      int nt = 64 + (wave - 6) * 2 + t;
      v8f acc = mm_tiles(sH2, HD, W3p, 68, nt, 2, lane);
      int n = nt * 16 + (lane & 15);
      float bb = b3[n];
      int o = n - 1024;
#pragma unroll
      for (int r = 0; r < 8; ++r) {
        int m = r + 8 * ((lane >> 4) & 1);
        float val = (acc[r] + bb) * sInvFn[m];
        gadd(&sc_sum[(size_t)sE0[m] * HD + o], val);
      }
    }
  }
  __syncthreads();
  // scatter per-edge f_c into per-node sums; bump counts
  for (int idx = tid; idx < 16 * 96; idx += 256) {
    int e = idx / 96, rem = idx % 96;
    gadd(&fc_sum[(size_t)sE0[e] * 96 + rem], ((float*)sFC)[idx]);
  }
  if (tid < 16) gadd(&cnt[sE0[tid]], 1.f);
}

// ---------------- node kernel: 16 nodes / block, 256 threads ----------------
__global__ __launch_bounds__(256) void k_node(
    const float* __restrict__ f, const float* __restrict__ s,
    const unsigned short* __restrict__ E2p,
    const float* __restrict__ b1, const float* __restrict__ b2, const float* __restrict__ b3,
    const unsigned short* __restrict__ S1p, const unsigned short* __restrict__ S2p,
    const unsigned short* __restrict__ S3p,
    const float* __restrict__ fc_sum, const float* __restrict__ sc_sum,
    const float* __restrict__ cnt,
    float* __restrict__ f_out, float* __restrict__ s_out) {
  __shared__ unsigned short sX[16 * SELF_KPAD];  // low region reused for 48x64 embed A-panel
  __shared__ float sTF[16][3][VD];
  __shared__ unsigned short sH1[16 * HD];
  __shared__ unsigned short sH2[16 * HD];
  __shared__ float sFO[16][3][VD];
  __shared__ float sInvFn[16];
  __shared__ float sSumSq[16];
  __shared__ float sRC[16];

  const int tid = threadIdx.x, lane = tid & 31, wave = tid >> 5;
  const int nbase = blockIdx.x * 16;

  if (tid < 16) {
    sRC[tid] = 1.f / fmaxf(cnt[nbase + tid], 1.f);
    sSumSq[tid] = 0.f;
  }
  for (int i = tid; i < 16 * 3 * VD; i += 256) ((float*)sFO)[i] = 0.f;
  __syncthreads();

  // stage embed A-panel: 48 rows x 64 cols = [f(32), fc_mean(32)]
  for (int idx = tid; idx < 48 * EMB2_KPAD; idx += 256) {
    int r = idx / EMB2_KPAD, k = idx % EMB2_KPAD, e = r / 3, i = r % 3, n = nbase + e;
    float v = (k < 32) ? f[(size_t)n * 96 + i * VD + k]
                       : fc_sum[(size_t)n * 96 + i * VD + (k - 32)] * sRC[e];
    sX[idx] = f2bf(v);
  }
  __syncthreads();

  // embed GEMM: [48,64] @ [64,32] -> sTF (f32)
  if (wave < 6) {
    int mtile = wave >> 1, nt = wave & 1;
    v8f acc = mm_tiles(sX + mtile * 16 * EMB2_KPAD, EMB2_KPAD, E2p, 2, nt, 2, lane);
    int c = nt * 16 + (lane & 15);
#pragma unroll
    for (int r = 0; r < 8; ++r) {
      int grow = mtile * 16 + r + 8 * ((lane >> 4) & 1);
      sTF[grow / 3][grow % 3][c] = acc[r];
    }
  }
  __syncthreads();

  // tf2s -> bf16 X2[:,0:1024] (+norm); scalar tail cols 1024..1151
  {
    int e = tid >> 4, base = (tid & 15) * 64;
    float ss = 0.f;
    for (int t = 0; t < 64; ++t) {
      int jk = base + t, j = jk >> 5, k = jk & 31;
      float v = sTF[e][0][j] * sTF[e][0][k] + sTF[e][1][j] * sTF[e][1][k] +
                sTF[e][2][j] * sTF[e][2][k];
      ss += v * v;
      sX[e * SELF_KPAD + jk] = f2bf(v);
    }
    atomicAdd(&sSumSq[e], ss);
  }
  for (int idx = tid; idx < 16 * 128; idx += 256) {
    int e = idx / 128, o = idx % 128, n = nbase + e;
    float v = (o < 64) ? s[(size_t)n * HD + o]
                       : sc_sum[(size_t)n * HD + (o - 64)] * sRC[e];
    sX[e * SELF_KPAD + 1024 + o] = f2bf(v);
  }
  __syncthreads();
  if (tid < 16) sInvFn[tid] = 1.f / (sqrtf(sSumSq[tid]) + 1.f);
  __syncthreads();

  if (wave < 4) {
    v8f acc = mm_tiles(sX, SELF_KPAD, S1p, 4, wave, SELF_KPAD / 32, lane);
    int n = wave * 16 + (lane & 15);
    float bb = b1[n];
#pragma unroll
    for (int r = 0; r < 8; ++r) {
      int m = r + 8 * ((lane >> 4) & 1);
      float h = acc[r] + bb;
      sH1[m * HD + n] = f2bf(h > 0.f ? h : 0.f);
    }
  }
  __syncthreads();
  if (wave < 4) {
    v8f acc = mm_tiles(sH1, HD, S2p, 4, wave, 2, lane);
    int n = wave * 16 + (lane & 15);
    float bb = b2[n];
#pragma unroll
    for (int r = 0; r < 8; ++r) {
      int m = r + 8 * ((lane >> 4) & 1);
      float h = acc[r] + bb;
      sH2[m * HD + n] = f2bf(h > 0.f ? h : 0.f);
    }
  }
  __syncthreads();

  if (wave < 6) {
    int h = wave & 1, slice = wave >> 1;
    int j0 = (slice == 0) ? 0 : (slice == 1) ? 11 : 22;
    int j1 = (slice == 0) ? 11 : (slice == 1) ? 22 : 32;
    float foacc[8][3];
#pragma unroll
    for (int r = 0; r < 8; ++r)
      for (int i = 0; i < 3; ++i) foacc[r][i] = 0.f;
    for (int j = j0; j < j1; ++j) {
      int nt = 2 * j + h;
      v8f acc = mm_tiles(sH2, HD, S3p, 68, nt, 2, lane);
      float bb = b3[nt * 16 + (lane & 15)];
#pragma unroll
      for (int r = 0; r < 8; ++r) {
        int m = r + 8 * ((lane >> 4) & 1);
        float val = (acc[r] + bb) * sInvFn[m];
        foacc[r][0] += sTF[m][0][j] * val;
        foacc[r][1] += sTF[m][1][j] * val;
        foacc[r][2] += sTF[m][2][j] * val;
      }
    }
    int k = h * 16 + (lane & 15);
#pragma unroll
    for (int r = 0; r < 8; ++r) {
      int m = r + 8 * ((lane >> 4) & 1);
      atomicAdd(&sFO[m][0][k], foacc[r][0]);
      atomicAdd(&sFO[m][1][k], foacc[r][1]);
      atomicAdd(&sFO[m][2][k], foacc[r][2]);
    }
  } else {
    for (int t = 0; t < 2; ++t) {
      int nt = 64 + (wave - 6) * 2 + t;
      v8f acc = mm_tiles(sH2, HD, S3p, 68, nt, 2, lane);
      int n = nt * 16 + (lane & 15);
      float bb = b3[n];
      int o = n - 1024;
#pragma unroll
      for (int r = 0; r < 8; ++r) {
        int m = r + 8 * ((lane >> 4) & 1);
        float val = (acc[r] + bb) * sInvFn[m];
        int node = nbase + m;
        s_out[(size_t)node * HD + o] = val + s[(size_t)node * HD + o];
      }
    }
  }
  __syncthreads();
  // f_out = tf @ C2 + f
  for (int idx = tid; idx < 16 * 96; idx += 256) {
    int e = idx / 96, rem = idx % 96;
    f_out[(size_t)(nbase + e) * 96 + rem] =
        ((float*)sFO)[idx] + f[(size_t)(nbase + e) * 96 + rem];
  }
}

extern "C" void kernel_launch(void* const* d_in, const int* in_sizes, int n_in,
                              void* d_out, int out_size, void* d_ws, size_t ws_size,
                              hipStream_t stream) {
  const float* f       = (const float*)d_in[0];
  const float* s       = (const float*)d_in[1];
  const float* edge_f  = (const float*)d_in[2];
  const float* edge_s  = (const float*)d_in[3];
  const float* Wemb1   = (const float*)d_in[4];
  const float* netW1   = (const float*)d_in[5];
  const float* netb1   = (const float*)d_in[6];
  const float* netW2   = (const float*)d_in[7];
  const float* netb2   = (const float*)d_in[8];
  const float* netW3   = (const float*)d_in[9];
  const float* netb3   = (const float*)d_in[10];
  const float* Wemb2   = (const float*)d_in[11];
  const float* selfW1  = (const float*)d_in[12];
  const float* selfb1  = (const float*)d_in[13];
  const float* selfW2  = (const float*)d_in[14];
  const float* selfb2  = (const float*)d_in[15];
  const float* selfW3  = (const float*)d_in[16];
  const float* selfb3  = (const float*)d_in[17];
  const int*   eidx    = (const int*)d_in[18];

  char* ws = (char*)d_ws;
  size_t off = 0;
  auto carve = [&](size_t bytes) {
    char* p = ws + off;
    off = (off + bytes + 255) & ~(size_t)255;
    return p;
  };
  unsigned short* W1p = (unsigned short*)carve((size_t)NET_KPAD * HD * 2);
  unsigned short* W2p = (unsigned short*)carve((size_t)HD * HD * 2);
  unsigned short* W3p = (unsigned short*)carve((size_t)HD * OUTD * 2);
  unsigned short* S1p = (unsigned short*)carve((size_t)SELF_KPAD * HD * 2);
  unsigned short* S2p = (unsigned short*)carve((size_t)HD * HD * 2);
  unsigned short* S3p = (unsigned short*)carve((size_t)HD * OUTD * 2);
  unsigned short* E1p = (unsigned short*)carve((size_t)EMB1_KPAD * VD * 2);
  unsigned short* E2p = (unsigned short*)carve((size_t)EMB2_KPAD * VD * 2);
  float* fc_sum = (float*)carve((size_t)NN * 96 * 4);
  float* sc_sum = (float*)carve((size_t)NN * HD * 4);
  float* cntp   = (float*)carve((size_t)NN * 4);

  k_zero<<<(NN * 96 + 255) / 256, 256, 0, stream>>>(fc_sum, NN * 96);
  k_zero<<<(NN * HD + 255) / 256, 256, 0, stream>>>(sc_sum, NN * HD);
  k_zero<<<(NN + 255) / 256, 256, 0, stream>>>(cntp, NN);

  auto pack = [&](const float* src, unsigned short* dst, int K, int KpadTiles, int Ncols) {
    int total = KpadTiles * (Ncols >> 4) * 512;
    k_pack<<<(total + 255) / 256, 256, 0, stream>>>(src, dst, K, KpadTiles, Ncols);
  };
  pack(netW1, W1p, 1220, NET_KPAD / 32, HD);
  pack(netW2, W2p, HD, 2, HD);
  pack(netW3, W3p, HD, 2, OUTD);
  pack(selfW1, S1p, SELF_KPAD, SELF_KPAD / 32, HD);
  pack(selfW2, S2p, HD, 2, HD);
  pack(selfW3, S3p, HD, 2, OUTD);
  pack(Wemb1, E1p, 65, EMB1_KPAD / 32, VD);
  pack(Wemb2, E2p, HD, EMB2_KPAD / 32, VD);

  k_edge<<<EE / 16, 256, 0, stream>>>(f, s, edge_f, edge_s, E1p, netb1, netb2, netb3,
                                      W1p, W2p, W3p, eidx, fc_sum, sc_sum, cntp);

  float* f_out = (float*)d_out;
  float* s_out = f_out + (size_t)NN * 96;
  k_node<<<NN / 16, 256, 0, stream>>>(f, s, E2p, selfb1, selfb2, selfb3,
                                      S1p, S2p, S3p, fc_sum, sc_sum, cntp, f_out, s_out);
}
